// NucleusMoEImageTransformerBlock_45792941310320
// MI455X (gfx1250) — compile-verified, matrix operand order checked by reading, not run on hardware
//
#include <hip/hip_runtime.h>

// ---------------- types ----------------
typedef __attribute__((ext_vector_type(16))) __bf16 v16bf;
typedef __attribute__((ext_vector_type(8)))  float  v8f;
typedef unsigned short u16;
typedef __attribute__((ext_vector_type(8)))  u16    v8u16;

union Frag16 { v16bf v; v8u16 h[2]; u16 s[16]; };

// ---------------- problem constants ----------------
constexpr int kBS  = 2;
constexpr int kS   = 1024;
constexpr int kT   = 512;
constexpr int kDIM = 1536;
constexpr int kH   = 12;
constexpr int kHD  = 128;
constexpr int kKV  = 4;
constexpr int kJD  = 3584;
constexpr int kE   = 16;
constexpr int kINNER = 1344;
constexpr int kCAP = 512;              // ceil(8*1024/16)
constexpr int kKTOT = kS + kT;         // 1536
constexpr int kNTOK = kBS * kS;        // 2048
constexpr float kEPS = 1e-6f;

__device__ inline u16 f2bf(float x) {
  unsigned b = __float_as_uint(x);
  unsigned r = (b + 0x7FFFu + ((b >> 16) & 1u)) >> 16;
  return (u16)r;
}

// low 32 bits of a generic pointer to LDS == LDS byte offset (aperture layout)
__device__ inline unsigned lds_off(const void* p) { return (unsigned)(size_t)p; }

__device__ inline float block_reduce_sum(float v, float* red) {
  int t = threadIdx.x;
  red[t] = v; __syncthreads();
  for (int s = blockDim.x >> 1; s > 0; s >>= 1) {
    if (t < s) red[t] += red[t + s];
    __syncthreads();
  }
  float r = red[0]; __syncthreads();
  return r;
}

// ---------------- generic WMMA GEMM: C[M,N] = A[M,K](bf16) @ W[N,K]^T(f32->bf16) ----------------
enum { EPI_F32 = 0, EPI_BF16 = 1, EPI_SCATTER = 2 };

template<int EPI>
__global__ __launch_bounds__(256) void gemm_bf16_kernel(
    const u16* __restrict__ A, const float* __restrict__ W,
    void* __restrict__ Cout, const float* __restrict__ bias,
    int M, int N, int K,
    const int* __restrict__ gidx, const float* __restrict__ gate)
{
  __shared__ __align__(16) u16 As[128 * 64];   // 16 KB
  __shared__ __align__(16) u16 Bs[64 * 64];    // 8 KB
  const int n0 = blockIdx.x * 64;
  const int m0 = blockIdx.y * 128;
  const int tid = threadIdx.x, wave = tid >> 5, lane = tid & 31;
  const int r15 = lane & 15, kh = (lane >> 4) & 1;

  v8f acc0{}, acc1{}, acc2{}, acc3{};

  const int ra = tid >> 1, ca = (tid & 1) * 32;   // A tile: 64 bytes / thread (async DMA)
  const int rb = tid >> 2, cb = (tid & 3) * 16;   // B tile: 16 f32 / thread (convert)
  const unsigned aoff = lds_off(As + ra * 64 + ca);

  for (int k0 = 0; k0 < K; k0 += 64) {
    // A tile 128x64 bf16: async global -> LDS (no VGPR round-trip)
    {
      const u16* asrc = A + (size_t)(m0 + ra) * K + k0 + ca;
      asm volatile(
          "global_load_async_to_lds_b128 %0, %1, off\n\t"
          "global_load_async_to_lds_b128 %0, %1, off offset:16\n\t"
          "global_load_async_to_lds_b128 %0, %1, off offset:32\n\t"
          "global_load_async_to_lds_b128 %0, %1, off offset:48"
          :: "v"(aoff), "v"(asrc) : "memory");
    }
    // B tile 64x64: f32 weights -> bf16 LDS (conversion must pass through VALU)
    {
      const float* src = W + (size_t)(n0 + rb) * K + k0 + cb;
      __builtin_prefetch(src + 64, 0, 3);
      u16* dst = Bs + rb * 64 + cb;
      #pragma unroll
      for (int i = 0; i < 16; ++i) dst[i] = f2bf(src[i]);
    }
    asm volatile("s_wait_asynccnt 0" ::: "memory");
    __syncthreads();
    #pragma unroll
    for (int ks = 0; ks < 64; ks += 32) {
      Frag16 af;
      { const u16* p = As + (wave * 16 + r15) * 64 + ks + kh * 8;
        af.h[0] = *(const v8u16*)p; af.h[1] = *(const v8u16*)(p + 16); }
      #pragma unroll
      for (int j = 0; j < 4; ++j) {
        Frag16 bf;
        { const u16* p = Bs + (j * 16 + r15) * 64 + ks + kh * 8;
          bf.h[0] = *(const v8u16*)p; bf.h[1] = *(const v8u16*)(p + 16); }
        v8f& a = (j == 0) ? acc0 : (j == 1) ? acc1 : (j == 2) ? acc2 : acc3;
        a = __builtin_amdgcn_wmma_f32_16x16x32_bf16(false, af.v, false, bf.v,
                                                    (short)0, a, false, false);
      }
    }
    __syncthreads();
  }
  #pragma unroll
  for (int j = 0; j < 4; ++j) {
    v8f a = (j == 0) ? acc0 : (j == 1) ? acc1 : (j == 2) ? acc2 : acc3;
    #pragma unroll
    for (int r = 0; r < 8; ++r) {
      int m = m0 + wave * 16 + r + kh * 8;
      int n = n0 + j * 16 + r15;
      float v = a[r];
      if (EPI == EPI_F32) {
        if (bias) v += bias[n];
        ((float*)Cout)[(size_t)m * N + n] = v;
      } else if (EPI == EPI_BF16) {
        if (bias) v += bias[n];
        ((u16*)Cout)[(size_t)m * N + n] = f2bf(v);
      } else {
        atomicAdd(((float*)Cout) + (size_t)gidx[m] * kDIM + n, v * gate[m]);
      }
    }
  }
}

// ---------------- flash attention (one wave per 16-query tile) ----------------
__global__ __launch_bounds__(32) void attn_kernel(
    const u16* __restrict__ qbf,   // [BS,S,H,HD]
    const u16* __restrict__ kbf,   // [BS,KTOT,KV,HD]
    const u16* __restrict__ vbf,   // [BS,KTOT,KV,HD]
    u16* __restrict__ obf)         // [BS,S,H,HD]
{
  __shared__ __align__(16) u16 Ptile[16 * 32];
  __shared__ __align__(16) u16 Vtile[32 * kHD];
  int blk = blockIdx.x;
  int qt = blk % (kS / 16);
  int tmp = blk / (kS / 16);
  int h = tmp % kH, b = tmp / kH;
  int kvh = h / (kH / kKV);
  int lane = threadIdx.x;
  int r15 = lane & 15, kh = lane >> 4;
  const float sc = 0.08838834764831845f;  // 1/sqrt(128)

  Frag16 qf[4];
  const u16* qbase = qbf + ((size_t)(b * kS + qt * 16 + r15) * kH + h) * kHD;
  #pragma unroll
  for (int kc = 0; kc < 4; ++kc) {
    qf[kc].h[0] = *(const v8u16*)(qbase + kc * 32 + kh * 8);
    qf[kc].h[1] = *(const v8u16*)(qbase + kc * 32 + 16 + kh * 8);
  }

  v8f o[8] = {};
  float rmax[8], rsum[8];
  #pragma unroll
  for (int r = 0; r < 8; ++r) { rmax[r] = -1e30f; rsum[r] = 0.f; }

  const unsigned voff = lds_off(Vtile + lane * kHD);

  for (int kb = 0; kb < kKTOT / 32; ++kb) {
    int key0 = kb * 32;
    // stage V block (32 keys x 128 feat) via async DMA; overlaps with QK^T WMMAs
    {
      const u16* vsrc = vbf + ((size_t)(b * kKTOT + key0 + lane) * kKV + kvh) * kHD;
      asm volatile(
          "global_load_async_to_lds_b128 %0, %1, off\n\t"
          "global_load_async_to_lds_b128 %0, %1, off offset:16\n\t"
          "global_load_async_to_lds_b128 %0, %1, off offset:32\n\t"
          "global_load_async_to_lds_b128 %0, %1, off offset:48\n\t"
          "global_load_async_to_lds_b128 %0, %1, off offset:64\n\t"
          "global_load_async_to_lds_b128 %0, %1, off offset:80\n\t"
          "global_load_async_to_lds_b128 %0, %1, off offset:96\n\t"
          "global_load_async_to_lds_b128 %0, %1, off offset:112\n\t"
          "global_load_async_to_lds_b128 %0, %1, off offset:128\n\t"
          "global_load_async_to_lds_b128 %0, %1, off offset:144\n\t"
          "global_load_async_to_lds_b128 %0, %1, off offset:160\n\t"
          "global_load_async_to_lds_b128 %0, %1, off offset:176\n\t"
          "global_load_async_to_lds_b128 %0, %1, off offset:192\n\t"
          "global_load_async_to_lds_b128 %0, %1, off offset:208\n\t"
          "global_load_async_to_lds_b128 %0, %1, off offset:224\n\t"
          "global_load_async_to_lds_b128 %0, %1, off offset:240"
          :: "v"(voff), "v"(vsrc) : "memory");
    }
    // scores: two 16x16 tiles over 32 keys (K fragments straight from global)
    v8f s0{}, s1{};
    #pragma unroll
    for (int kc = 0; kc < 4; ++kc) {
      Frag16 b0, b1;
      const u16* k0p = kbf + ((size_t)(b * kKTOT + key0 + r15) * kKV + kvh) * kHD + kc * 32;
      const u16* k1p = kbf + ((size_t)(b * kKTOT + key0 + 16 + r15) * kKV + kvh) * kHD + kc * 32;
      b0.h[0] = *(const v8u16*)(k0p + kh * 8); b0.h[1] = *(const v8u16*)(k0p + 16 + kh * 8);
      b1.h[0] = *(const v8u16*)(k1p + kh * 8); b1.h[1] = *(const v8u16*)(k1p + 16 + kh * 8);
      s0 = __builtin_amdgcn_wmma_f32_16x16x32_bf16(false, qf[kc].v, false, b0.v, (short)0, s0, false, false);
      s1 = __builtin_amdgcn_wmma_f32_16x16x32_bf16(false, qf[kc].v, false, b1.v, (short)0, s1, false, false);
    }
    asm volatile("s_wait_asynccnt 0" ::: "memory");
    __syncthreads();
    // online softmax; lane owns rows M=r+8*kh, col r15 (and r15+16)
    #pragma unroll
    for (int r = 0; r < 8; ++r) {
      float v0 = s0[r] * sc, v1 = s1[r] * sc;
      float m = fmaxf(v0, v1);
      #pragma unroll
      for (int mk = 1; mk < 16; mk <<= 1) m = fmaxf(m, __shfl_xor(m, mk, 32));
      float mnew = fmaxf(rmax[r], m);
      float alpha = __expf(rmax[r] - mnew);
      float p0 = __expf(v0 - mnew), p1 = __expf(v1 - mnew);
      float ps = p0 + p1;
      #pragma unroll
      for (int mk = 1; mk < 16; mk <<= 1) ps += __shfl_xor(ps, mk, 32);
      rsum[r] = rsum[r] * alpha + ps;
      rmax[r] = mnew;
      #pragma unroll
      for (int fc = 0; fc < 8; ++fc) o[fc][r] *= alpha;
      int M = r + 8 * kh;
      Ptile[M * 32 + r15]      = f2bf(p0);
      Ptile[M * 32 + 16 + r15] = f2bf(p1);
    }
    __syncthreads();
    Frag16 pf;
    { const u16* pp = Ptile + r15 * 32 + kh * 8;
      pf.h[0] = *(const v8u16*)pp; pf.h[1] = *(const v8u16*)(pp + 16); }
    #pragma unroll
    for (int fc = 0; fc < 8; ++fc) {
      Frag16 vf;
      #pragma unroll
      for (int i = 0; i < 8; ++i) {
        vf.s[i]     = Vtile[(kh * 8 + i) * kHD + fc * 16 + r15];
        vf.s[8 + i] = Vtile[(16 + kh * 8 + i) * kHD + fc * 16 + r15];
      }
      o[fc] = __builtin_amdgcn_wmma_f32_16x16x32_bf16(false, pf.v, false, vf.v, (short)0, o[fc], false, false);
    }
    __syncthreads();
  }
  #pragma unroll
  for (int fc = 0; fc < 8; ++fc)
    #pragma unroll
    for (int r = 0; r < 8; ++r) {
      int M = r + 8 * kh;
      float val = o[fc][r] / rsum[r];
      obf[((size_t)(b * kS + qt * 16 + M) * kH + h) * kHD + fc * 16 + r15] = f2bf(val);
    }
}

// ---------------- elementwise / small kernels ----------------
__global__ void f32_to_bf16_kernel(const float* __restrict__ in, u16* __restrict__ out, size_t n) {
  size_t i = (size_t)blockIdx.x * blockDim.x + threadIdx.x;
  if (i < n) out[i] = f2bf(in[i]);
}

__global__ void silu_kernel(const float* __restrict__ in, float* __restrict__ out, int n) {
  int i = blockIdx.x * blockDim.x + threadIdx.x;
  if (i < n) { float x = in[i]; out[i] = x / (1.f + __expf(-x)); }
}

__global__ void mod_kernel(const float* __restrict__ ts, const float* __restrict__ Wm,
                           const float* __restrict__ bm, float* __restrict__ mod) {
  int i = blockIdx.x * blockDim.x + threadIdx.x;           // [0, 2*6144)
  int b = i / (4 * kDIM), n = i % (4 * kDIM);
  const float* w = Wm + (size_t)n * kDIM;
  const float* x = ts + b * kDIM;
  float acc = bm[n];
  for (int d = 0; d < kDIM; ++d) acc += x[d] * w[d];
  mod[i] = acc;
}

// LN over rows of DIM; writes bf16 modulated copy and optional raw f32 copy
__global__ __launch_bounds__(256) void ln_mod_kernel(
    const float* __restrict__ x, const float* __restrict__ mod,
    u16* __restrict__ out_bf, float* __restrict__ out_f32, int scale_off)
{
  __shared__ float red[256];
  int row = blockIdx.x, t = threadIdx.x, b = row / kS;
  const float* xr = x + (size_t)row * kDIM;
  float s = 0.f;
  for (int d = t; d < kDIM; d += 256) s += xr[d];
  float mean = block_reduce_sum(s, red) / kDIM;
  float v = 0.f;
  for (int d = t; d < kDIM; d += 256) { float dd = xr[d] - mean; v += dd * dd; }
  float inv = rsqrtf(block_reduce_sum(v, red) / kDIM + kEPS);
  for (int d = t; d < kDIM; d += 256) {
    float nx = (xr[d] - mean) * inv;
    if (out_f32) out_f32[(size_t)row * kDIM + d] = nx;
    float sc = 1.f + mod[(size_t)b * 4 * kDIM + scale_off + d];
    out_bf[(size_t)row * kDIM + d] = f2bf(nx * sc);
  }
}

// RMS over rows of HD=128 (q path: layout-preserving)
__global__ __launch_bounds__(128) void rms_rows_kernel(
    const float* __restrict__ in, const float* __restrict__ w, u16* __restrict__ out)
{
  __shared__ float red[128];
  int row = blockIdx.x, t = threadIdx.x;
  float x = in[(size_t)row * kHD + t];
  float ss = block_reduce_sum(x * x, red);
  float r = rsqrtf(ss / kHD + kEPS);
  out[(size_t)row * kHD + t] = f2bf(x * r * w[t]);
}

// K/V path: optional RMS + write into concatenated [BS,KTOT,KV,HD] layout
__global__ __launch_bounds__(128) void kv_norm_kernel(
    const float* __restrict__ in, const float* __restrict__ w, u16* __restrict__ out,
    int ntok, int base, int do_rms)
{
  __shared__ float red[128];
  int row = blockIdx.x, t = threadIdx.x;       // row in [0, BS*ntok*KV)
  int kv = row % kKV, tok = row / kKV;
  int b = tok / ntok, s = tok % ntok;
  float x = in[(size_t)row * kHD + t];
  float val = x;
  if (do_rms) {
    float ss = block_reduce_sum(x * x, red);
    val = x * rsqrtf(ss / kHD + kEPS) * w[t];
  }
  out[(((size_t)b * kKTOT + base + s) * kKV + kv) * kHD + t] = f2bf(val);
}

__global__ void resid_kernel(const float* __restrict__ base, const float* __restrict__ delta,
                             const float* __restrict__ mod, int gate_off, float* __restrict__ out) {
  size_t i = (size_t)blockIdx.x * blockDim.x + threadIdx.x;
  if (i >= (size_t)kNTOK * kDIM) return;
  int d = i % kDIM;
  int b = (int)(i / ((size_t)kS * kDIM));
  float g = mod[(size_t)b * 4 * kDIM + gate_off + d];
  g = fminf(2.f, fmaxf(-2.f, g));
  out[i] = base[i] + tanhf(g) * delta[i];
}

__global__ __launch_bounds__(256) void router_kernel(
    const float* __restrict__ temb, const float* __restrict__ normed2,
    const float* __restrict__ Wg, float* __restrict__ scores)
{
  __shared__ float red[256];
  __shared__ float lg[kE];
  int tok = blockIdx.x, t = threadIdx.x, b = tok / kS;
  for (int e = 0; e < kE; ++e) {
    float p = 0.f;
    const float* wg = Wg + (size_t)e * 2 * kDIM;
    for (int d = t; d < kDIM; d += 256)
      p += temb[b * kDIM + d] * wg[d] + normed2[(size_t)tok * kDIM + d] * wg[kDIM + d];
    float s = block_reduce_sum(p, red);
    if (t == 0) lg[e] = s;
  }
  __syncthreads();
  if (t == 0) {
    float m = -1e30f;
    for (int e = 0; e < kE; ++e) m = fmaxf(m, lg[e]);
    float z = 0.f;
    for (int e = 0; e < kE; ++e) { lg[e] = __expf(lg[e] - m); z += lg[e]; }
    for (int e = 0; e < kE; ++e) scores[(size_t)tok * kE + e] = lg[e] / z;
  }
}

// per-(b,e) top-512 of 1024 via bitonic sort; accumulates segment sums
__global__ __launch_bounds__(512) void topk_kernel(
    const float* __restrict__ scores, int* __restrict__ gidx,
    float* __restrict__ gating0, float* __restrict__ sums)
{
  __shared__ float val[kS];
  __shared__ int idx[kS];
  int be = blockIdx.x, b = be / kE, e = be % kE, t = threadIdx.x;
  for (int i = t; i < kS; i += 512) {
    val[i] = scores[((size_t)b * kS + i) * kE + e];
    idx[i] = i;
  }
  __syncthreads();
  for (int k = 2; k <= kS; k <<= 1) {
    for (int j = k >> 1; j > 0; j >>= 1) {
      int i = 2 * j * (t / j) + (t % j);
      int p = i + j;
      bool descRegion = ((i & k) == 0);
      float vi = val[i], vp = val[p];
      bool doswap = descRegion ? (vi < vp) : (vi > vp);
      if (doswap) {
        val[i] = vp; val[p] = vi;
        int ti = idx[i]; idx[i] = idx[p]; idx[p] = ti;
      }
      __syncthreads();
    }
  }
  int s = idx[t];
  float g = val[t];
  int flat = (e * kBS + b) * kCAP + t;
  gidx[flat] = b * kS + s;
  gating0[flat] = g;
  atomicAdd(&sums[b * kS + s], g);
}

__global__ void gate_norm_kernel(const float* __restrict__ gating0, const int* __restrict__ gidx,
                                 const float* __restrict__ sums, float* __restrict__ gate) {
  int i = blockIdx.x * blockDim.x + threadIdx.x;
  if (i < kE * kBS * kCAP) gate[i] = gating0[i] / (sums[gidx[i]] + 1e-12f) * 2.5f;
}

__global__ __launch_bounds__(256) void gather_kernel(
    const u16* __restrict__ mod2_bf, const int* __restrict__ gidx, u16* __restrict__ routed) {
  int row = blockIdx.x;
  const u16* src = mod2_bf + (size_t)gidx[row] * kDIM;
  u16* dst = routed + (size_t)row * kDIM;
  for (int d = threadIdx.x; d < kDIM; d += 256) dst[d] = src[d];
}

__global__ void swiglu_kernel(const float* __restrict__ h, u16* __restrict__ out, size_t n) {
  size_t i = (size_t)blockIdx.x * blockDim.x + threadIdx.x;   // n = M*INNER
  if (i >= n) return;
  size_t m = i / kINNER, c = i % kINNER;
  float a = h[m * 2 * kINNER + c];
  float g = h[m * 2 * kINNER + kINNER + c];
  out[i] = f2bf(a * (g / (1.f + __expf(-g))));
}

// ---------------- launcher ----------------
extern "C" void kernel_launch(void* const* d_in, const int* in_sizes, int n_in,
                              void* d_out, int out_size, void* d_ws, size_t ws_size,
                              hipStream_t stream) {
  (void)in_sizes; (void)n_in; (void)out_size; (void)ws_size;
  const float* hidden = (const float*)d_in[0];
  const float* enc    = (const float*)d_in[1];
  const float* temb   = (const float*)d_in[2];
  const float* W_mod  = (const float*)d_in[3];
  const float* b_mod  = (const float*)d_in[4];
  const float* W_enc  = (const float*)d_in[5];
  const float* b_enc  = (const float*)d_in[6];
  const float* Wq     = (const float*)d_in[7];
  const float* Wk     = (const float*)d_in[8];
  const float* Wv     = (const float*)d_in[9];
  const float* Wak    = (const float*)d_in[10];
  const float* Wav    = (const float*)d_in[11];
  const float* nq_w   = (const float*)d_in[12];
  const float* nk_w   = (const float*)d_in[13];
  const float* nak_w  = (const float*)d_in[14];
  const float* Wo     = (const float*)d_in[15];
  const float* Wg     = (const float*)d_in[16];
  const float* We_in  = (const float*)d_in[17];
  const float* We_out = (const float*)d_in[18];
  const float* Ws_in  = (const float*)d_in[19];
  const float* Ws_out = (const float*)d_in[20];
  float* out = (float*)d_out;

  char* wp = (char*)d_ws;
  auto alloc = [&](size_t bytes) -> void* {
    void* r = (void*)wp;
    wp += (bytes + 255) & ~(size_t)255;
    return r;
  };
  float* temb_silu = (float*)alloc((size_t)kBS * kDIM * 4);
  float* modv      = (float*)alloc((size_t)kBS * 4 * kDIM * 4);
  u16*   enc_bf    = (u16*)alloc((size_t)kBS * kT * kJD * 2);
  u16*   ctx_bf    = (u16*)alloc((size_t)kBS * kT * kDIM * 2);
  u16*   mod1_bf   = (u16*)alloc((size_t)kNTOK * kDIM * 2);
  float* q_f       = (float*)alloc((size_t)kNTOK * kDIM * 4);
  u16*   q_bf      = (u16*)alloc((size_t)kNTOK * kDIM * 2);
  float* kvtmp     = (float*)alloc((size_t)kNTOK * kKV * kHD * 4);
  u16*   k_bf      = (u16*)alloc((size_t)kBS * kKTOT * kKV * kHD * 2);
  u16*   v_bf      = (u16*)alloc((size_t)kBS * kKTOT * kKV * kHD * 2);
  u16*   attn_bf   = (u16*)alloc((size_t)kNTOK * kDIM * 2);
  float* proj_f    = (float*)alloc((size_t)kNTOK * kDIM * 4);
  float* hs        = (float*)alloc((size_t)kNTOK * kDIM * 4);
  float* normed2   = (float*)alloc((size_t)kNTOK * kDIM * 4);
  u16*   mod2_bf   = (u16*)alloc((size_t)kNTOK * kDIM * 2);
  float* scores    = (float*)alloc((size_t)kNTOK * kE * 4);
  int*   gidx      = (int*)alloc((size_t)kE * kBS * kCAP * 4);
  float* gating0   = (float*)alloc((size_t)kE * kBS * kCAP * 4);
  float* gate      = (float*)alloc((size_t)kE * kBS * kCAP * 4);
  float* sums      = (float*)alloc((size_t)kNTOK * 4);
  u16*   routed_bf = (u16*)alloc((size_t)kE * kBS * kCAP * kDIM * 2);
  float* shh_f     = (float*)alloc((size_t)kNTOK * 2 * kINNER * 4);
  u16*   shact_bf  = (u16*)alloc((size_t)kNTOK * kINNER * 2);
  float* hexp_f    = (float*)alloc((size_t)(kBS * kCAP) * 2 * kINNER * 4);
  u16*   hact_bf   = (u16*)alloc((size_t)(kBS * kCAP) * kINNER * 2);
  float* moe_out   = (float*)alloc((size_t)kNTOK * kDIM * 4);

  // 1) encoder -> bf16
  {
    size_t n = (size_t)kBS * kT * kJD;
    f32_to_bf16_kernel<<<(unsigned)((n + 255) / 256), 256, 0, stream>>>(enc, enc_bf, n);
  }
  // 2) modulation
  silu_kernel<<<(kBS * kDIM + 255) / 256, 256, 0, stream>>>(temb, temb_silu, kBS * kDIM);
  mod_kernel<<<(kBS * 4 * kDIM + 255) / 256, 256, 0, stream>>>(temb_silu, W_mod, b_mod, modv);
  // 3) context = enc @ W_enc^T + b_enc   [1024,3584]x[1536,3584]
  gemm_bf16_kernel<EPI_BF16><<<dim3(kDIM / 64, (kBS * kT) / 128), 256, 0, stream>>>(
      enc_bf, W_enc, ctx_bf, b_enc, kBS * kT, kDIM, kJD, nullptr, nullptr);
  // 4) LN1 * scale1
  ln_mod_kernel<<<kNTOK, 256, 0, stream>>>(hidden, modv, mod1_bf, nullptr, 0);
  // 5) Q proj + head RMS
  gemm_bf16_kernel<EPI_F32><<<dim3(kDIM / 64, kNTOK / 128), 256, 0, stream>>>(
      mod1_bf, Wq, q_f, nullptr, kNTOK, kDIM, kDIM, nullptr, nullptr);
  rms_rows_kernel<<<kNTOK * kH, 128, 0, stream>>>(q_f, nq_w, q_bf);
  // 6) K_img
  gemm_bf16_kernel<EPI_F32><<<dim3((kKV * kHD) / 64, kNTOK / 128), 256, 0, stream>>>(
      mod1_bf, Wk, kvtmp, nullptr, kNTOK, kKV * kHD, kDIM, nullptr, nullptr);
  kv_norm_kernel<<<kNTOK * kKV, 128, 0, stream>>>(kvtmp, nk_w, k_bf, kS, 0, 1);
  // 7) V_img
  gemm_bf16_kernel<EPI_F32><<<dim3((kKV * kHD) / 64, kNTOK / 128), 256, 0, stream>>>(
      mod1_bf, Wv, kvtmp, nullptr, kNTOK, kKV * kHD, kDIM, nullptr, nullptr);
  kv_norm_kernel<<<kNTOK * kKV, 128, 0, stream>>>(kvtmp, nk_w, v_bf, kS, 0, 0);
  // 8) K_txt
  gemm_bf16_kernel<EPI_F32><<<dim3((kKV * kHD) / 64, (kBS * kT) / 128), 256, 0, stream>>>(
      ctx_bf, Wak, kvtmp, nullptr, kBS * kT, kKV * kHD, kDIM, nullptr, nullptr);
  kv_norm_kernel<<<kBS * kT * kKV, 128, 0, stream>>>(kvtmp, nak_w, k_bf, kT, kS, 1);
  // 9) V_txt
  gemm_bf16_kernel<EPI_F32><<<dim3((kKV * kHD) / 64, (kBS * kT) / 128), 256, 0, stream>>>(
      ctx_bf, Wav, kvtmp, nullptr, kBS * kT, kKV * kHD, kDIM, nullptr, nullptr);
  kv_norm_kernel<<<kBS * kT * kKV, 128, 0, stream>>>(kvtmp, nak_w, v_bf, kT, kS, 0);
  // 10) attention
  attn_kernel<<<kBS * kH * (kS / 16), 32, 0, stream>>>(q_bf, k_bf, v_bf, attn_bf);
  // 11) output projection + gated residual
  gemm_bf16_kernel<EPI_F32><<<dim3(kDIM / 64, kNTOK / 128), 256, 0, stream>>>(
      attn_bf, Wo, proj_f, nullptr, kNTOK, kDIM, kDIM, nullptr, nullptr);
  resid_kernel<<<(unsigned)(((size_t)kNTOK * kDIM + 255) / 256), 256, 0, stream>>>(
      hidden, proj_f, modv, kDIM, hs);
  // 12) LN2 (raw f32 for router + modulated bf16 for experts)
  ln_mod_kernel<<<kNTOK, 256, 0, stream>>>(hs, modv, mod2_bf, normed2, 2 * kDIM);
  // 13) router + expert-choice top-k
  router_kernel<<<kNTOK, 256, 0, stream>>>(temb, normed2, Wg, scores);
  hipMemsetAsync(sums, 0, (size_t)kNTOK * 4, stream);
  topk_kernel<<<kBS * kE, 512, 0, stream>>>(scores, gidx, gating0, sums);
  gate_norm_kernel<<<(kE * kBS * kCAP + 255) / 256, 256, 0, stream>>>(gating0, gidx, sums, gate);
  gather_kernel<<<kE * kBS * kCAP, 256, 0, stream>>>(mod2_bf, gidx, routed_bf);
  // 14) shared expert -> moe_out (plain store initializes buffer)
  gemm_bf16_kernel<EPI_F32><<<dim3((2 * kINNER) / 64, kNTOK / 128), 256, 0, stream>>>(
      mod2_bf, Ws_in, shh_f, nullptr, kNTOK, 2 * kINNER, kDIM, nullptr, nullptr);
  {
    size_t n = (size_t)kNTOK * kINNER;
    swiglu_kernel<<<(unsigned)((n + 255) / 256), 256, 0, stream>>>(shh_f, shact_bf, n);
  }
  gemm_bf16_kernel<EPI_F32><<<dim3(kDIM / 64, kNTOK / 128), 256, 0, stream>>>(
      shact_bf, Ws_out, moe_out, nullptr, kNTOK, kDIM, kINNER, nullptr, nullptr);
  // 15) routed experts: gemm1 -> swiglu -> gemm2 (gated scatter-add)
  const int TPE = kBS * kCAP;  // 1024 tokens per expert
  for (int e = 0; e < kE; ++e) {
    const u16* Ae = routed_bf + (size_t)e * TPE * kDIM;
    const float* W1 = We_in + (size_t)e * 2 * kINNER * kDIM;
    const float* W2 = We_out + (size_t)e * kDIM * kINNER;
    gemm_bf16_kernel<EPI_F32><<<dim3((2 * kINNER) / 64, TPE / 128), 256, 0, stream>>>(
        Ae, W1, hexp_f, nullptr, TPE, 2 * kINNER, kDIM, nullptr, nullptr);
    size_t n = (size_t)TPE * kINNER;
    swiglu_kernel<<<(unsigned)((n + 255) / 256), 256, 0, stream>>>(hexp_f, hact_bf, n);
    gemm_bf16_kernel<EPI_SCATTER><<<dim3(kDIM / 64, TPE / 128), 256, 0, stream>>>(
        hact_bf, W2, moe_out, nullptr, TPE, kDIM, kINNER, gidx + e * TPE, gate + e * TPE);
  }
  // 16) final gated residual -> d_out
  resid_kernel<<<(unsigned)(((size_t)kNTOK * kDIM + 255) / 256), 256, 0, stream>>>(
      hs, moe_out, modv, 3 * kDIM, out);
}
